// MultiDomainNet_20873541058960
// MI455X (gfx1250) — compile-verified, matrix operand order checked by reading, not run on hardware
//
#include <hip/hip_runtime.h>
#include <hip/hip_bf16.h>

// ---- problem constants ----
#define NB   8
#define ND   16
#define NAA  20
#define NL   512
#define NC2  4096
#define NC   2048   // C2/2
#define NF   12

typedef __attribute__((ext_vector_type(16))) __bf16 v16bf;
typedef __attribute__((ext_vector_type(8)))  float  v8f;

// Workspace layout (bytes)
#define OFF_ADJB   ((size_t)0)                       // bf16 [D][C2][L]      64 MB
#define OFF_ADJET  ((size_t)67108864)                // bf16 [D][L][C]       32 MB
#define OFF_DEG    ((size_t)100663296)               // f32  [D][L]          32 KB
#define OFF_AW     ((size_t)100696064)               // bf16 [B*D][16][1024]  4 MB

// LDS layout (dynamic) — padded row strides (+8 bf16 = 16B): stride 516/1028 dwords
// => 16 lanes x b128 cover all 64 banks exactly once (conflict-free).
#define A1_STRIDE   1032
#define CV_STRIDE   2056
#define LDS_A1      0
#define LDS_CONV    33024
#define LDS_DEG     98816
#define LDS_FMAX    100864
#define LDS_PART    100928
#define LDS_TOTAL   101952

// v16bf WMMA fragment: per-lane two contiguous 8-element (16B) K-chunks,
// per ISA 16-bit A layout (K[kb..kb+7] -> vgpr0..3, K[kb+16..kb+23] -> vgpr4..7).
__device__ __forceinline__ v16bf frag_from(const __bf16* p, int kb) {
  union { v16bf v; uint4 q[2]; } u;
  u.q[0] = *reinterpret_cast<const uint4*>(p + kb);
  u.q[1] = *reinterpret_cast<const uint4*>(p + kb + 16);
  return u.v;
}

#define WMMA_BF16(A, Bf, C) \
  __builtin_amdgcn_wmma_f32_16x16x32_bf16(false, (A), false, (Bf), (short)0, (C), false, false)

// ---------- K0: zero degree accumulator ----------
__global__ void zero_deg(float* __restrict__ deg) {
  deg[blockIdx.x * 256 + threadIdx.x] = 0.f;
}

// ---------- K1: adjs f32 -> bf16 copy + degree sums, then coalesced transposed-even copy ----------
__global__ void prep_adj(const float* __restrict__ adjs, void* adjB_, void* adjET_,
                         float* __restrict__ deg) {
  __bf16* adjB  = (__bf16*)adjB_;
  __bf16* adjET = (__bf16*)adjET_;
  const int d = blockIdx.x;
  const int cblk = blockIdx.y;      // 64 contact rows per block
  const int tid = threadIdx.x;      // 256 threads
  const size_t dbase = (size_t)d * NC2 * NL;

  // Pass 1: bf16 copy (coalesced) + per-l degree partials
  float acc0 = 0.f, acc1 = 0.f;
  for (int r = 0; r < 64; ++r) {
    const int c = cblk * 64 + r;
    const float* row = adjs + dbase + (size_t)c * NL;
    const float v0 = row[tid];
    const float v1 = row[tid + 256];
    adjB[dbase + (size_t)c * NL + tid]       = (__bf16)v0;
    adjB[dbase + (size_t)c * NL + tid + 256] = (__bf16)v1;
    acc0 += v0; acc1 += v1;
  }
  atomicAdd(&deg[d * NL + tid],       acc0);
  atomicAdd(&deg[d * NL + tid + 256], acc1);

  // Pass 2: transposed even rows, coalesced writes (reads are L2-hot from pass 1)
  for (int idx = tid; idx < 32 * NL; idx += 256) {
    const int l = idx >> 5;          // residue
    const int j = idx & 31;          // local even-row index
    const int c = cblk * 64 + 2 * j;
    const float v = adjs[dbase + (size_t)c * NL + l];
    adjET[((size_t)d * NL + l) * NC + cblk * 32 + j] = (__bf16)v;
  }
}

// ---------- K2: fold conv weights into A operand: Ã[f,k,s] = Σ_n w[f,n,k]·aa[b,n,d,s] ----------
__global__ void prep_A(const float* __restrict__ aa, const float* __restrict__ conv_w,
                       void* A_) {
  __bf16* Aw = (__bf16*)A_;
  const int bd = blockIdx.x;               // b*ND + d
  const int b = bd / ND, d = bd % ND;
  __shared__ float wsh[NF * NAA * 2];
  for (int i = threadIdx.x; i < NF * NAA * 2; i += blockDim.x) wsh[i] = conv_w[i];
  __syncthreads();
  __bf16* Arow = Aw + (size_t)bd * 16 * 1024;   // [16 rows][K=1024], rows 12..15 zero
  for (int s = threadIdx.x; s < NL; s += blockDim.x) {
    float a[NAA];
    #pragma unroll
    for (int n = 0; n < NAA; ++n)
      a[n] = aa[((size_t)b * NAA + n) * (ND * NL) + d * NL + s];
    for (int f = 0; f < NF; ++f) {
      float s0 = 0.f, s1 = 0.f;
      #pragma unroll
      for (int n = 0; n < NAA; ++n) {
        s0 += wsh[(f * NAA + n) * 2 + 0] * a[n];
        s1 += wsh[(f * NAA + n) * 2 + 1] * a[n];
      }
      Arow[f * 1024 + s]       = (__bf16)s0;   // k=0 half of fused K
      Arow[f * 1024 + 512 + s] = (__bf16)s1;   // k=1 half
    }
    #pragma unroll
    for (int f = NF; f < 16; ++f) {
      Arow[f * 1024 + s]       = (__bf16)0.f;
      Arow[f * 1024 + 512 + s] = (__bf16)0.f;
    }
  }
}

// ---------- K3: fused WMMA GEMM-GEMM-reduce per (b,d) ----------
__global__ void __launch_bounds__(256, 1)
fused_main(const void* adjB_, const void* adjET_, const float* __restrict__ deg,
           const void* A_, const float* __restrict__ lin_w, float* __restrict__ out) {
  extern __shared__ char smem[];
  __bf16* A1    = (__bf16*)(smem + LDS_A1);     // [16][1032] bf16 (1024 used)
  __bf16* convB = (__bf16*)(smem + LDS_CONV);   // [16][2056] bf16 (2048 used)
  float*  out2  = (float*)(smem + LDS_A1);      // overlay after phase A: [16][512] f32
  float*  degs  = (float*)(smem + LDS_DEG);     // [512]
  float*  fmaxb = (float*)(smem + LDS_FMAX);    // [16]
  float*  part  = (float*)(smem + LDS_PART);    // [16][16]

  const int bd = blockIdx.x;
  const int b = bd / ND, d = bd % ND;
  const int tid  = threadIdx.x;
  const int wid  = tid >> 5;       // 8 waves (wave32)
  const int lane = tid & 31;
  const int hl   = lane >> 4;      // K-half selector per ISA A-layout
  const int mn   = lane & 15;      // row (A) / column (B,D)

  const __bf16* adjB_d  = (const __bf16*)adjB_  + (size_t)d * NC2 * NL;
  const __bf16* adjET_d = (const __bf16*)adjET_ + (size_t)d * NL * NC;

  // Stage A operand (32KB) + degrees into LDS
  {
    const uint4* src = (const uint4*)((const __bf16*)A_ + (size_t)bd * 16 * 1024);
    for (int i = tid; i < 2048; i += 256) {
      const int row = i >> 7, cq = i & 127;          // 128 uint4 per 1024-elem row
      *reinterpret_cast<uint4*>(A1 + row * A1_STRIDE + cq * 8) = src[i];
    }
    for (int i = tid; i < NL; i += 256) degs[i] = deg[d * NL + i];
  }
  __syncthreads();

  const __bf16* a1row = A1 + mn * A1_STRIDE;

  // ---- Phase A: conv[16][2048] = A1[16][1024] x adjB-cols[1024][2048] ----
  // 4 passes of 4 register-blocked N-tiles; double-buffered fragment loads.
  for (int pass = 0; pass < 4; ++pass) {
    const int nt0 = wid * 16 + pass * 4;
    const __bf16* bc0 = adjB_d + (size_t)(2 * ((nt0 + 0) * 16 + mn)) * NL;
    const __bf16* bc1 = adjB_d + (size_t)(2 * ((nt0 + 1) * 16 + mn)) * NL;
    const __bf16* bc2 = adjB_d + (size_t)(2 * ((nt0 + 2) * 16 + mn)) * NL;
    const __bf16* bc3 = adjB_d + (size_t)(2 * ((nt0 + 3) * 16 + mn)) * NL;
    __builtin_prefetch(bc0 + 4 * 16 * 2 * NL, 0, 1);   // next pass
    v8f acc0 = {}, acc1 = {}, acc2 = {}, acc3 = {};
    int kb = hl * 8;
    v16bf a_c  = frag_from(a1row, kb);
    v16bf b0_c = frag_from(bc0, kb), b1_c = frag_from(bc1, kb);
    v16bf b2_c = frag_from(bc2, kb), b3_c = frag_from(bc3, kb);
    #pragma unroll 4
    for (int step = 0; step < 31; ++step) {
      const int kbn = kb + 32;
      v16bf a_n  = frag_from(a1row, kbn);
      v16bf b0_n = frag_from(bc0, kbn), b1_n = frag_from(bc1, kbn);
      v16bf b2_n = frag_from(bc2, kbn), b3_n = frag_from(bc3, kbn);
      acc0 = WMMA_BF16(a_c, b0_c, acc0);
      acc1 = WMMA_BF16(a_c, b1_c, acc1);
      acc2 = WMMA_BF16(a_c, b2_c, acc2);
      acc3 = WMMA_BF16(a_c, b3_c, acc3);
      a_c = a_n; b0_c = b0_n; b1_c = b1_n; b2_c = b2_n; b3_c = b3_n;
      kb = kbn;
    }
    acc0 = WMMA_BF16(a_c, b0_c, acc0);
    acc1 = WMMA_BF16(a_c, b1_c, acc1);
    acc2 = WMMA_BF16(a_c, b2_c, acc2);
    acc3 = WMMA_BF16(a_c, b3_c, acc3);
    // D layout: lanes<16 hold rows 0..7, lanes>=16 rows 8..15; column = tile*16+mn
    #pragma unroll
    for (int r = 0; r < 8; ++r) {
      const int mrow = (r + hl * 8) * CV_STRIDE;
      convB[mrow + (nt0 + 0) * 16 + mn] = (__bf16)acc0[r];
      convB[mrow + (nt0 + 1) * 16 + mn] = (__bf16)acc1[r];
      convB[mrow + (nt0 + 2) * 16 + mn] = (__bf16)acc2[r];
      convB[mrow + (nt0 + 3) * 16 + mn] = (__bf16)acc3[r];
    }
  }
  __syncthreads();

  // ---- Phase B: out2[16][512] = convB[16][2048] x adjET-cols[2048][512] ----
  {
    const int nt0 = wid * 4;
    const __bf16* cvrow = convB + mn * CV_STRIDE;
    const __bf16* br0 = adjET_d + (size_t)((nt0 + 0) * 16 + mn) * NC;
    const __bf16* br1 = adjET_d + (size_t)((nt0 + 1) * 16 + mn) * NC;
    const __bf16* br2 = adjET_d + (size_t)((nt0 + 2) * 16 + mn) * NC;
    const __bf16* br3 = adjET_d + (size_t)((nt0 + 3) * 16 + mn) * NC;
    v8f acc0 = {}, acc1 = {}, acc2 = {}, acc3 = {};
    int kb = hl * 8;
    v16bf a_c  = frag_from(cvrow, kb);
    v16bf b0_c = frag_from(br0, kb), b1_c = frag_from(br1, kb);
    v16bf b2_c = frag_from(br2, kb), b3_c = frag_from(br3, kb);
    #pragma unroll 4
    for (int step = 0; step < 63; ++step) {
      const int kbn = kb + 32;
      v16bf a_n  = frag_from(cvrow, kbn);
      v16bf b0_n = frag_from(br0, kbn), b1_n = frag_from(br1, kbn);
      v16bf b2_n = frag_from(br2, kbn), b3_n = frag_from(br3, kbn);
      acc0 = WMMA_BF16(a_c, b0_c, acc0);
      acc1 = WMMA_BF16(a_c, b1_c, acc1);
      acc2 = WMMA_BF16(a_c, b2_c, acc2);
      acc3 = WMMA_BF16(a_c, b3_c, acc3);
      a_c = a_n; b0_c = b0_n; b1_c = b1_n; b2_c = b2_n; b3_c = b3_n;
      kb = kbn;
    }
    acc0 = WMMA_BF16(a_c, b0_c, acc0);
    acc1 = WMMA_BF16(a_c, b1_c, acc1);
    acc2 = WMMA_BF16(a_c, b2_c, acc2);
    acc3 = WMMA_BF16(a_c, b3_c, acc3);
    const int l0 = nt0 * 16 + mn;
    const float d0 = degs[l0], d1 = degs[l0 + 16], d2 = degs[l0 + 32], d3 = degs[l0 + 48];
    #pragma unroll
    for (int r = 0; r < 8; ++r) {
      const int mrow = (r + hl * 8) * NL;
      out2[mrow + l0]      = acc0[r] / d0;
      out2[mrow + l0 + 16] = acc1[r] / d1;
      out2[mrow + l0 + 32] = acc2[r] / d2;
      out2[mrow + l0 + 48] = acc3[r] / d3;
    }
  }
  __syncthreads();

  // ---- Phase C: rowwise max over residues, dot with lin_w, sigmoid ----
  {
    const int row = tid >> 4, col = tid & 15;       // 16 threads per output row
    float m = -3.402823466e38f;
    const float* prow = out2 + row * NL;
    for (int l = col; l < NL; l += 16) m = fmaxf(m, prow[l]);
    part[row * 16 + col] = m;
  }
  __syncthreads();
  if (tid < 16) {
    float m = -3.402823466e38f;
    #pragma unroll
    for (int j = 0; j < 16; ++j) m = fmaxf(m, part[tid * 16 + j]);
    fmaxb[tid] = m;
  }
  __syncthreads();
  if (tid == 0) {
    float s = 0.f;
    #pragma unroll
    for (int f = 0; f < NF; ++f) s += fmaxb[f] * lin_w[f];
    out[d * NB + b] = 1.f / (1.f + __expf(-s));   // scores are [D*B], d-major
  }
}

extern "C" void kernel_launch(void* const* d_in, const int* in_sizes, int n_in,
                              void* d_out, int out_size, void* d_ws, size_t ws_size,
                              hipStream_t stream) {
  (void)in_sizes; (void)n_in; (void)out_size; (void)ws_size;
  const float* aa     = (const float*)d_in[0];
  const float* adjs   = (const float*)d_in[1];
  const float* conv_w = (const float*)d_in[2];
  const float* lin_w  = (const float*)d_in[3];
  float* out = (float*)d_out;

  char* ws = (char*)d_ws;
  void*  adjB  = ws + OFF_ADJB;
  void*  adjET = ws + OFF_ADJET;
  float* deg   = (float*)(ws + OFF_DEG);
  void*  Aw    = ws + OFF_AW;

  zero_deg<<<32, 256, 0, stream>>>(deg);
  prep_adj<<<dim3(ND, NC2 / 64), 256, 0, stream>>>(adjs, adjB, adjET, deg);
  prep_A<<<NB * ND, 256, 0, stream>>>(aa, conv_w, Aw);
  fused_main<<<NB * ND, 256, LDS_TOTAL, stream>>>(adjB, adjET, deg, Aw, lin_w, out);
}
